// MetaPathAdapterS_39633958208187
// MI455X (gfx1250) — matrix operand order repaired and checked by reading, not compiled, not run on hardware
//
#include <hip/hip_runtime.h>
#include <math.h>

typedef __attribute__((ext_vector_type(16))) _Float16 v16h;
typedef __attribute__((ext_vector_type(8)))  float    v8f;

#define N_A 50000
#define N_P 50000
#define N_C 25000

union PairPack { _Float16 h[2]; unsigned u; };

// ---------------- zero ----------------
__global__ void zero_f32(float* __restrict__ p, long n) {
    long i = (long)blockIdx.x * blockDim.x + threadIdx.x;
    long stride = (long)gridDim.x * blockDim.x;
    for (; i < n; i += stride) p[i] = 0.0f;
}

// Layout helpers for the 16-bit WMMA fragments (ISA 7.12.2).
// A (16x32): lane L = m + 16*hi holds elements e=0..15 with
//   kk = e + 8*hi (e<8), kk = 8 + e + 8*hi (e>=8); pairs (e,e+1) <-> (kk,kk+1), kk even.
// B (32x16): lane L = n + 16*hi holds elements e=0..15 with the same kk order as A.

// ---------------- GEMM: OUT[m, out_off:out_off+128] = X[m,0:K] @ W[K,128] ----------------
// Block = 128 threads (4 waves). Block computes a 64-row x 128-col tile:
// each wave owns one 16-row tile and all 8 N-tiles (8 v8f accumulators).
// B fragments are double-buffered so ds_load_b128 for step t+1 overlaps WMMA t.
__global__ __launch_bounds__(128) void gemm_f16_wmma(
    const float* __restrict__ X, const float* __restrict__ W,
    float* __restrict__ OUT, int M, int K, int out_stride, int out_off)
{
    __shared__ __align__(32) _Float16 ldsA[4][32][16];   // 4 KB: per-wave packed A fragments
    __shared__ __align__(32) _Float16 ldsB[8][32][16];   // 8 KB: packed B fragments per N-tile
    const int tid  = threadIdx.x;
    const int wave = tid >> 5;
    const int L    = tid & 31;
    const int n    = L & 15;
    const int hi   = L >> 4;
    const int rowbase = blockIdx.x * 64 + wave * 16;

    v8f zero = {};
    v8f acc[8];
    #pragma unroll
    for (int t = 0; t < 8; ++t) acc[t] = zero;

    for (int k0 = 0; k0 < K; k0 += 32) {
        // ---- stage this wave's A tile: 16 rows x 32 k = 256 pairs / 32 lanes
        for (int idx = L; idx < 256; idx += 32) {
            int m  = idx >> 4;
            int kk = (idx & 15) * 2;
            int gr = rowbase + m; if (gr >= M) gr = M - 1;
            const float* src = X + (long)gr * K + k0 + kk;
            if (k0 + 32 < K) __builtin_prefetch(src + 32, 0, 0);
            float2 xv = *(const float2*)src;
            int h  = (kk >> 3) & 1;
            int kl = kk - 8 * h;
            int e  = (kl < 8) ? kl : (kl - 8);
            PairPack pk; pk.h[0] = (_Float16)xv.x; pk.h[1] = (_Float16)xv.y;
            *(unsigned*)&ldsA[wave][m + 16 * h][e] = pk.u;
        }
        // ---- stage B tile cooperatively: 32 k x 128 n = 2048 pairs / 128 threads
        for (int idx = tid; idx < 2048; idx += 128) {
            int kk = (idx >> 7) * 2;
            int c  = idx & 127;
            const float* src = W + (long)(k0 + kk) * 128 + c;
            float b0 = src[0];
            float b1 = src[128];
            int h = (kk >> 3) & 1;
            int e = (kk & 7) + ((kk >> 4) << 3);
            PairPack pk; pk.h[0] = (_Float16)b0; pk.h[1] = (_Float16)b1;
            *(unsigned*)&ldsB[c >> 4][(c & 15) + 16 * h][e] = pk.u;
        }
        __syncthreads();

        v16h afrag = *(const v16h*)ldsA[wave][L];
        v16h bcur  = *(const v16h*)ldsB[0][L];
        #pragma unroll
        for (int t = 0; t < 8; ++t) {
            v16h bnext = bcur;
            if (t < 7) bnext = *(const v16h*)ldsB[t + 1][L];
            acc[t] = __builtin_amdgcn_wmma_f32_16x16x32_f16(
                false, afrag, false, bcur, (short)0, acc[t], false, false);
            bcur = bnext;
        }
        __syncthreads();
    }

    // C/D layout: VGPR r, lanes 0-15 -> M=r, N=lane; lanes 16-31 -> M=r+8, N=lane-16.
    float* o = OUT + (long)(rowbase + hi * 8) * out_stride + out_off + n;
    if (rowbase + 16 <= M) {
        #pragma unroll
        for (int r = 0; r < 8; ++r) {
            #pragma unroll
            for (int t = 0; t < 8; ++t)
                o[t * 16] = acc[t][r];
            o += out_stride;
        }
    } else {
        for (int r = 0; r < 8; ++r) {
            if (rowbase + hi * 8 + r < M) {
                #pragma unroll
                for (int t = 0; t < 8; ++t)
                    o[t * 16] = acc[t][r];
            }
            o += out_stride;
        }
    }
}

// ---------------- scatter-add (segment sum + degree count) ----------------
__global__ void scatter_add_mean(const float* __restrict__ H,
                                 const int* __restrict__ rows,
                                 const int* __restrict__ cols, int E,
                                 float* __restrict__ outb, int out_stride, int out_off,
                                 float* __restrict__ deg)
{
    int gw = (blockIdx.x * blockDim.x + threadIdx.x) >> 5;
    int L  = threadIdx.x & 31;
    if (gw >= E) return;
    int r = rows[gw], c = cols[gw];
    const float4 v = ((const float4*)(H + (long)r * 128))[L];
    float* dst = outb + (long)c * out_stride + out_off + L * 4;
    atomicAdd(dst + 0, v.x);
    atomicAdd(dst + 1, v.y);
    atomicAdd(dst + 2, v.z);
    atomicAdd(dst + 3, v.w);
    if (L == 0) atomicAdd(deg + c, 1.0f);
}

// ---------------- divide by clamped degree ----------------
__global__ void normalize_mean(float* __restrict__ Hs,
                               const float* __restrict__ deg1,
                               const float* __restrict__ deg2)
{
    int i = blockIdx.x * blockDim.x + threadIdx.x;
    if (i >= N_A * 128) return;
    int nn = i >> 7, d = i & 127;
    float i1 = 1.0f / fmaxf(deg1[nn], 1.0f);
    float i2 = 1.0f / fmaxf(deg2[nn], 1.0f);
    Hs[(long)nn * 384 + d]       *= i1;
    Hs[(long)nn * 384 + 128 + d] *= i2;
}

// ---------------- semantic attention scores ----------------
// scores[row] = sum_c tanh( (Hs[row,:] @ Wsem)[c] + b[c] ) * q[c],  rows = N_A*3
__global__ __launch_bounds__(128) void semantic_scores_wmma(
    const float* __restrict__ Hs,     // [R,128] flat
    const float* __restrict__ Wsem,   // [128,128]
    const float* __restrict__ bias,   // [128]
    const float* __restrict__ q,      // [128]
    float* __restrict__ scores, int R)
{
    __shared__ __align__(32) _Float16 ldsW[4][8][32][16];   // 32 KB packed B fragments
    __shared__ __align__(32) _Float16 ldsAs[4][4][32][16];  // 16 KB packed A fragments per wave
    const int tid  = threadIdx.x;
    const int wave = tid >> 5;
    const int L    = tid & 31;
    const int n    = L & 15;
    const int hi   = L >> 4;
    const int rowbase = (blockIdx.x * 4 + wave) * 16;

    // stage Wsem: 8192 k-pairs, packed
    for (int idx = tid; idx < 8192; idx += 128) {
        int kk = (idx >> 7) * 2;            // 0..126 even
        int c  = idx & 127;
        const float* src = Wsem + (long)kk * 128 + c;
        float b0 = src[0];
        float b1 = src[128];
        int ks = kk >> 5;
        int kl = kk & 31;
        int h  = (kl >> 3) & 1;
        int e  = (kl & 7) + ((kl >> 4) << 3);
        PairPack pk; pk.h[0] = (_Float16)b0; pk.h[1] = (_Float16)b1;
        *(unsigned*)&ldsW[ks][c >> 4][(c & 15) + 16 * h][e] = pk.u;
    }
    // each wave stages its own 16x128 A tile: 1024 pairs (float2 loads)
    for (int idx = L; idx < 1024; idx += 32) {
        int r  = idx >> 6;
        int kk = (idx & 63) * 2;            // 0..126 even
        int gr = rowbase + r; if (gr >= R) gr = R - 1;
        float2 xv = *(const float2*)(Hs + (long)gr * 128 + kk);
        int ks  = kk >> 5;
        int kkl = kk & 31;
        int h   = (kkl >> 3) & 1;
        int kl  = kkl - 8 * h;
        int e   = (kl < 8) ? kl : (kl - 8);
        PairPack pk; pk.h[0] = (_Float16)xv.x; pk.h[1] = (_Float16)xv.y;
        *(unsigned*)&ldsAs[wave][ks][r + 16 * h][e] = pk.u;
    }
    __syncthreads();

    v8f zero = {};
    v8f acc[8];
    #pragma unroll
    for (int t = 0; t < 8; ++t) acc[t] = zero;

    // software-pipelined fragment loads: load (ks,t)+1 before WMMA (ks,t)
    v16h afrag = *(const v16h*)ldsAs[wave][0][L];
    v16h bcur  = *(const v16h*)ldsW[0][0][L];
    #pragma unroll
    for (int ks = 0; ks < 4; ++ks) {
        v16h anext = afrag;
        if (ks < 3) anext = *(const v16h*)ldsAs[wave][ks + 1][L];
        #pragma unroll
        for (int t = 0; t < 8; ++t) {
            v16h bnext = bcur;
            if (t < 7)       bnext = *(const v16h*)ldsW[ks][t + 1][L];
            else if (ks < 3) bnext = *(const v16h*)ldsW[ks + 1][0][L];
            acc[t] = __builtin_amdgcn_wmma_f32_16x16x32_f16(
                false, afrag, false, bcur, (short)0, acc[t], false, false);
            bcur = bnext;
        }
        afrag = anext;
    }

    float bv[8], qv[8];
    #pragma unroll
    for (int t = 0; t < 8; ++t) { bv[t] = bias[t * 16 + n]; qv[t] = q[t * 16 + n]; }

    #pragma unroll
    for (int r = 0; r < 8; ++r) {
        float s = 0.0f;
        #pragma unroll
        for (int t = 0; t < 8; ++t)
            s += tanhf(acc[t][r] + bv[t]) * qv[t];
        s += __shfl_xor(s, 1, 32);
        s += __shfl_xor(s, 2, 32);
        s += __shfl_xor(s, 4, 32);
        s += __shfl_xor(s, 8, 32);
        if (n == 0) {
            int row = rowbase + hi * 8 + r;
            if (row < R) scores[row] = s;
        }
    }
}

// ---------------- softmax over M=3 ----------------
__global__ void softmax3(float* __restrict__ alpha, int n_nodes) {
    int i = blockIdx.x * blockDim.x + threadIdx.x;
    if (i >= n_nodes) return;
    float a = alpha[i * 3 + 0], b = alpha[i * 3 + 1], c = alpha[i * 3 + 2];
    float m = fmaxf(a, fmaxf(b, c));
    float ea = expf(a - m), eb = expf(b - m), ec = expf(c - m);
    float inv = 1.0f / (ea + eb + ec);
    alpha[i * 3 + 0] = ea * inv;
    alpha[i * 3 + 1] = eb * inv;
    alpha[i * 3 + 2] = ec * inv;
}

extern "C" void kernel_launch(void* const* d_in, const int* in_sizes, int n_in,
                              void* d_out, int out_size, void* d_ws, size_t ws_size,
                              hipStream_t stream)
{
    const float* x_A     = (const float*)d_in[0];
    const float* x_P     = (const float*)d_in[1];
    const float* x_C     = (const float*)d_in[2];
    const float* W_A     = (const float*)d_in[3];
    const float* W_P     = (const float*)d_in[4];
    const float* W_C     = (const float*)d_in[5];
    const float* W_sem_w = (const float*)d_in[6];
    const float* W_sem_b = (const float*)d_in[7];
    const float* sem_q   = (const float*)d_in[8];
    const int*   ei_pa   = (const int*)d_in[10];
    const int*   ei_ca   = (const int*)d_in[12];
    const int E_pa = in_sizes[10] / 2;
    const int E_ca = in_sizes[12] / 2;

    float* Hs    = (float*)d_out;                 // [N_A, 3, 128]
    float* alpha = Hs + (long)N_A * 3 * 128;      // [N_A, 3]

    char* ws = (char*)d_ws;
    float* hP   = (float*)ws; ws += (size_t)N_P * 128 * sizeof(float);
    float* hC   = (float*)ws; ws += (size_t)N_C * 128 * sizeof(float);
    float* deg1 = (float*)ws; ws += (size_t)N_A * sizeof(float);
    float* deg2 = (float*)ws;

    // 1) zero accumulation targets (d_out/d_ws are poisoned by harness)
    zero_f32<<<2048, 256, 0, stream>>>(Hs, (long)N_A * 384);
    zero_f32<<<256, 256, 0, stream>>>(deg1, N_A);
    zero_f32<<<256, 256, 0, stream>>>(deg2, N_A);

    // 2) projections (hA directly into Hs slot 2); 64 rows per block
    gemm_f16_wmma<<<(N_A + 63) / 64, 128, 0, stream>>>(x_A, W_A, Hs, N_A, 256, 384, 256);
    gemm_f16_wmma<<<(N_P + 63) / 64, 128, 0, stream>>>(x_P, W_P, hP, N_P, 256, 128, 0);
    gemm_f16_wmma<<<(N_C + 63) / 64, 128, 0, stream>>>(x_C, W_C, hC, N_C, 128, 128, 0);

    // 3) scatter segment-sum + degree (slot 0 from hP via ei_pa, slot 1 from hC via ei_ca)
    scatter_add_mean<<<(E_pa * 32 + 255) / 256, 256, 0, stream>>>(
        hP, ei_pa, ei_pa + E_pa, E_pa, Hs, 384, 0, deg1);
    scatter_add_mean<<<(E_ca * 32 + 255) / 256, 256, 0, stream>>>(
        hC, ei_ca, ei_ca + E_ca, E_ca, Hs, 384, 128, deg2);

    // 4) divide by clamped in-degree
    normalize_mean<<<(N_A * 128 + 255) / 256, 256, 0, stream>>>(Hs, deg1, deg2);

    // 5) semantic scores (raw) into alpha region
    const int R = N_A * 3;
    const int rtiles = (R + 15) / 16;
    semantic_scores_wmma<<<(rtiles + 3) / 4, 128, 0, stream>>>(
        Hs, W_sem_w, W_sem_b, sem_q, alpha, R);

    // 6) softmax over the 3 metapaths
    softmax3<<<(N_A + 255) / 256, 256, 0, stream>>>(alpha, N_A);
}
